// SpatialEncoder3d_74165495267609
// MI455X (gfx1250) — matrix operand order repaired and checked by reading, not compiled, not run on hardware
//
#include <hip/hip_runtime.h>
#include <hip/hip_bf16.h>
#include <math.h>

typedef __attribute__((ext_vector_type(16))) _Float16 v16h;
typedef __attribute__((ext_vector_type(8)))  float    v8f;

namespace {
constexpr int kB = 16;
constexpr int kN = 256;
constexpr int kK = 128;
constexpr int kH = 8;
}

struct __align__(16) GParam { float mean, ivar, coef, pad; };

__device__ __forceinline__ float gelu_exact(float x) {
    return 0.5f * x * (1.0f + erff(x * 0.70710678118654752440f));
}

// Compile-time scheduling fence: keeps LDS loads next to their uses so the
// compiler cannot hoist/cluster them into giant live ranges (which spilled).
__device__ __forceinline__ void sched_fence() {
    asm volatile("" ::: "memory");
}

__global__ __launch_bounds__(256, 4) void spatial3d_wmma_kernel(
    const float* __restrict__ coords,
    const float* __restrict__ emb,
    const float* __restrict__ w1,
    const float* __restrict__ b1,
    const float* __restrict__ W2,
    const float* __restrict__ b2,
    const float* __restrict__ W3,
    const float* __restrict__ b3,
    const int*   __restrict__ num_nodes,
    float*       __restrict__ out)
{
    // W2 staged as f16 in WMMA B-fragment order: [frag = t*4+s][lane][elem]
    __shared__ __align__(32) _Float16 lds_w2[32 * 32 * 16];   // 32 KB
    __shared__ GParam lds_gp[kK];                             // 2 KB, one b128/elem
    __shared__ float  lds_b2[kK];
    __shared__ float  lds_w3[kH * kK];
    __shared__ float  lds_b3[kH];

    const int tid  = threadIdx.x;
    const int lane = tid & 31;
    const int wave = tid >> 5;
    const int g    = lane & 15;
    const bool hi  = lane >= 16;

    // ---- one-time staging per workgroup ----
    // B-fragment layout for V_WMMA_F32_16X16X32_F16 (32x16 B):
    //   lane n (0-15): col n,   K = (e<8 ? e : e+8)      (e = 0..15)
    //   lane n (16-31): col n-16, K = 8 + (e<8 ? e : e+8)
    #pragma unroll 4
    for (int nrep = 0; nrep < 64; ++nrep) {
        int idx  = tid + nrep * 256;
        int frag = idx >> 9;          // 512 halves per fragment
        int rem  = idx & 511;
        int fl   = rem >> 4;          // lane within fragment
        int e    = rem & 15;          // element within v16h
        int t    = frag >> 2;         // N-tile 0..7
        int s    = frag & 3;          // K-step 0..3
        int col  = t * 16 + (fl & 15);
        int k    = 32 * s + ((fl < 16) ? 0 : 8) + ((e < 8) ? e : e + 8);
        lds_w2[idx] = (_Float16)W2[col * kK + k];
    }
    if (tid < kK) {
        float iv = 1.0f / fabsf(emb[tid * 2 + 1]);
        GParam p;
        p.mean = emb[tid * 2 + 0];
        p.ivar = iv;
        p.coef = -0.3989422804014327f * iv;   // 1/(-sqrt(2*pi)*var)
        p.pad  = 0.0f;
        lds_gp[tid] = p;
        lds_b2[tid] = b2[tid];
    }
    for (int idx = tid; idx < kH * kK; idx += 256) lds_w3[idx] = W3[idx];
    if (tid < kH) lds_b3[tid] = b3[tid];
    __syncthreads();

    const int bb = blockIdx.x >> 8;
    const int i  = blockIdx.x & 255;
    const int nn = num_nodes[bb];
    const float w1v = w1[0];
    const float b1v = b1[0];
    const float cix = coords[(bb * kN + i) * 3 + 0];
    const float ciy = coords[(bb * kN + i) * 3 + 1];
    const float ciz = coords[(bb * kN + i) * 3 + 2];

    // Per-lane register caches for the 128->8 head projection (loaded once;
    // immune to the memory fences below since they live in VGPRs).
    // D-layout: lane holds channel n = t*16 + g for tiles t=0..7.
    float w3reg[kH][8];
    #pragma unroll
    for (int h = 0; h < kH; ++h)
        #pragma unroll
        for (int t = 0; t < 8; ++t)
            w3reg[h][t] = lds_w3[h * kK + t * 16 + g];
    float b2v[8];
    #pragma unroll
    for (int t = 0; t < 8; ++t) b2v[t] = lds_b2[t * 16 + g];
    const float b3v = lds_b3[(lane & 15) >> 1];   // head this lane will store

    const int kbaseA = hi ? 8 : 0;   // A-fragment K sub-block offset per lane half

    #pragma unroll 1
    for (int iter = 0; iter < 2; ++iter) {
        // Barrier: blocks LICM/CSE of LDS loads across the two j-tiles.
        __syncthreads();

        const int j0 = (wave + 8 * iter) * 16;   // 16-pair tile: j in [j0, j0+16)

        // ---- distance + scalar affine for this lane's A-row (row m = g) ----
        const int jrow = j0 + g;
        const float* cj = &coords[(bb * kN + jrow) * 3];
        const float dx = cix - cj[0];
        const float dy = ciy - cj[1];
        const float dz = ciz - cj[2];
        const float ss = dx * dx + dy * dy + dz * dz;
        const float dist = (ss > 0.0f) ? sqrtf(ss) : 0.0f;
        const float scaled = w1v * dist + b1v;

        v8f acc[8] = {};

        // ---- 16x128 @ 128x128 GEMM: one A-fragment live at a time ----
        // A-layout (16x32 f16): lane L: row M=L%16, K = 32s + base + (e<8 ? e : e+8)
        #pragma unroll
        for (int s = 0; s < 4; ++s) {
            sched_fence();            // keep this step's param loads local
            v16h a;
            #pragma unroll
            for (int e = 0; e < 16; ++e) {
                const int k = 32 * s + kbaseA + ((e < 8) ? e : e + 8);
                const GParam p = lds_gp[k];
                const float z = (scaled - p.mean) * p.ivar;
                a[e] = (_Float16)(__expf(-0.5f * z * z) * p.coef);
            }
            #pragma unroll
            for (int t = 0; t < 8; ++t) {
                sched_fence();        // exactly one B-fragment live at a time
                const v16h bf = *(const v16h*)&lds_w2[(((t * 4 + s) * 32) + lane) * 16];
                acc[t] = __builtin_amdgcn_wmma_f32_16x16x32_f16(
                            /*neg_a=*/false, a, /*neg_b=*/false, bf,
                            /*c_mod=*/(short)0, acc[t],
                            /*reuse_a=*/false, /*reuse_b=*/false);
            }
        }

        // ---- +b2, exact GELU, 128->8 head projection, mask, store ----
        // D-layout: VGPR r, lane: pair m = r + (hi?8:0), channel n = t*16 + g
        #pragma unroll
        for (int r = 0; r < 8; ++r) {
            float x[8];
            #pragma unroll
            for (int t = 0; t < 8; ++t)
                x[t] = gelu_exact(acc[t][r] + b2v[t]);

            float v[kH];
            #pragma unroll
            for (int h = 0; h < kH; ++h) {
                float sum = 0.0f;
                #pragma unroll
                for (int t = 0; t < 8; ++t)
                    sum = fmaf(x[t], w3reg[h][t], sum);
                v[h] = sum;
            }

            // Folding reduction over the 16-lane group: each step halves the
            // h-set a lane carries while summing across the lane pair.
            // After 4 steps lane g holds the full sum for h = g>>1.
            #pragma unroll
            for (int q = 0; q < 4; ++q) {
                const float keep = (g & 8) ? v[q + 4] : v[q];
                const float send = (g & 8) ? v[q]     : v[q + 4];
                v[q] = keep + __shfl_xor(send, 8, 32);
            }
            #pragma unroll
            for (int q = 0; q < 2; ++q) {
                const float keep = (g & 4) ? v[q + 2] : v[q];
                const float send = (g & 4) ? v[q]     : v[q + 2];
                v[q] = keep + __shfl_xor(send, 4, 32);
            }
            {
                const float keep = (g & 2) ? v[1] : v[0];
                const float send = (g & 2) ? v[0] : v[1];
                v[0] = keep + __shfl_xor(send, 2, 32);
            }
            v[0] += __shfl_xor(v[0], 1, 32);

            const int m  = r + (hi ? 8 : 0);
            const int jw = j0 + m;
            if ((g & 1) == 0) {
                const int h = g >> 1;
                const bool valid = (i < nn) && (jw < nn);
                const float val = valid ? (v[0] + b3v) : -INFINITY;
                out[(((bb * kN + i) * kN) + jw) * kH + h] = val;
            }
        }
    }
}

extern "C" void kernel_launch(void* const* d_in, const int* in_sizes, int n_in,
                              void* d_out, int out_size, void* d_ws, size_t ws_size,
                              hipStream_t stream) {
    const float* coords = (const float*)d_in[0];
    const float* emb    = (const float*)d_in[1];
    const float* w1     = (const float*)d_in[2];
    const float* b1     = (const float*)d_in[3];
    const float* W2     = (const float*)d_in[4];
    const float* b2     = (const float*)d_in[5];
    const float* W3     = (const float*)d_in[6];
    const float* b3     = (const float*)d_in[7];
    const int*   nnz    = (const int*)d_in[8];
    float* out = (float*)d_out;

    dim3 grid(kB * kN);   // one (b, i) row per block; 8 waves x 2 j-tiles each
    dim3 block(256);
    spatial3d_wmma_kernel<<<grid, block, 0, stream>>>(
        coords, emb, w1, b1, W2, b2, W3, b3, nnz, out);
}
